// GAT_893353198189
// MI455X (gfx1250) — matrix operand order repaired and checked by reading, not compiled, not run on hardware
//
#include <hip/hip_runtime.h>
#include <hip/hip_bf16.h>
#include <math.h>

#define NNODES 50000
#define NEDGES 800000
#define ETOT   (NNODES + NEDGES)   // 850000 edges incl. self loops
#define FIN    512
#define FHID   512                 // HEADS*HID
#define HEADS  8
#define HIDC   64
#define NOUT   40
#define NEG_SLOPE 0.2f

typedef __bf16 bf16;
typedef __attribute__((ext_vector_type(16))) __bf16 v16bf;
typedef __attribute__((ext_vector_type(8)))  float  v8f;

struct frag32 { uint4 lo, hi; };   // 32 bytes == one v16bf, element order lo||hi

// ---------------- elementwise helpers ----------------

__global__ void cvt_bf16_kernel(const float* __restrict__ in, bf16* __restrict__ out, long n) {
    long i = (long)blockIdx.x * blockDim.x + threadIdx.x;
    if (i < n) out[i] = (bf16)in[i];
}

__global__ void fill_f32_kernel(float* __restrict__ p, float v, long n) {
    long i = (long)blockIdx.x * blockDim.x + threadIdx.x;
    if (i < n) p[i] = v;
}

// Wt[n*K + k] = bf16(W[k*Nf + n]); rows n in [Nf, NfPad) are zero padding
__global__ void transpose_bf16_kernel(const float* __restrict__ W, bf16* __restrict__ Wt,
                                      int K, int Nf, int NfPad) {
    long i = (long)blockIdx.x * blockDim.x + threadIdx.x;
    long total = (long)NfPad * K;
    if (i >= total) return;
    int n = (int)(i / K), k = (int)(i % K);
    float v = (n < Nf) ? W[(long)k * Nf + n] : 0.f;
    Wt[i] = (bf16)v;
}

// ---------------- WMMA bf16 GEMM: C[M,Nf] = A[M,K] x Bt[N,K]^T ----------------
// block = 16 rows of C, 8 waves; wave w owns N-tiles w, w+8, ..., w+(TPW-1)*8.
// TPW is compile-time so accumulators stay in statically-indexed VGPRs (no v_movrel).
template <int TPW>
__global__ __launch_bounds__(256)
void gemm_wmma_kernel(const bf16* __restrict__ A, const bf16* __restrict__ Bt,
                      float* __restrict__ C, int K, int Nf, int NT) {
    const int lane  = threadIdx.x & 31;
    const int wave  = threadIdx.x >> 5;
    const int khalf = lane >> 4;     // which K-half this lane holds
    const int l15   = lane & 15;     // M (for A) / N (for B,C) within tile
    const int m0    = blockIdx.x * 16;

    // wave-uniform participation check (keeps EXEC all-ones for WMMA)
    if (wave + (TPW - 1) * 8 >= NT) return;

    v8f acc[TPW] = {};
    const bf16* Arow = A + (long)(m0 + l15) * K;
    const int ksteps = K >> 5;

    for (int ks = 0; ks < ksteps; ks++) {
        // A fragment: 16-bit A 16x32 layout -> element i: K = (i/8)*16 + khalf*8 + i%8
        const bf16* Ap = Arow + ks * 32 + khalf * 8;
        frag32 fa;
        fa.lo = *(const uint4*)(Ap);
        fa.hi = *(const uint4*)(Ap + 16);
        v16bf a = __builtin_bit_cast(v16bf, fa);
        __builtin_prefetch(Ap + 32, 0, 0);          // next K-step A line

#pragma unroll
        for (int j = 0; j < TPW; j++) {
            // B fragment: 16-bit B 32x16 layout -> element i: K = (i/8)*16 + khalf*8 + i%8, N = l15
            const int n = (wave + j * 8) * 16 + l15;
            const bf16* Bp = Bt + (long)n * K + ks * 32 + khalf * 8;
            frag32 fb;
            fb.lo = *(const uint4*)(Bp);
            fb.hi = *(const uint4*)(Bp + 16);
            v16bf b = __builtin_bit_cast(v16bf, fb);
            __builtin_prefetch(Bp + 32, 0, 0);      // next K-step B line

            acc[j] = __builtin_amdgcn_wmma_f32_16x16x32_bf16(
                false, a, false, b, (short)0, acc[j], false, false);
        }
    }

    // f32 C/D layout: VGPR r -> row r + 8*khalf, column = l15
#pragma unroll
    for (int j = 0; j < TPW; j++) {
        int n = (wave + j * 8) * 16 + l15;
        if (n < Nf) {
#pragma unroll
            for (int r = 0; r < 8; r++)
                C[(long)(m0 + r + 8 * khalf) * Nf + n] = acc[j][r];
        }
    }
}

// ---------------- attention coefficients per node ----------------

__global__ void alpha_kernel(const float* __restrict__ h, const float* __restrict__ as,
                             const float* __restrict__ ad, float* __restrict__ osrc,
                             float* __restrict__ odst, int H, int C) {
    long idx = (long)blockIdx.x * blockDim.x + threadIdx.x;   // n*H + head
    if (idx >= (long)NNODES * H) return;
    int n = (int)(idx / H), hh = (int)(idx % H);
    const float* row = h + (long)n * H * C + (long)hh * C;
    float s = 0.f, d = 0.f;
    for (int c = 0; c < C; c++) {
        float v = row[c];
        s = fmaf(v, as[hh * C + c], s);
        d = fmaf(v, ad[hh * C + c], d);
    }
    osrc[idx] = s; odst[idx] = d;
}

// ---------------- segment softmax over edges ----------------

__device__ inline void atomicMaxFloat(float* addr, float val) {
    if (val >= 0.f) atomicMax((int*)addr, __float_as_int(val));
    else            atomicMin((unsigned int*)addr, (unsigned int)__float_as_int(val));
}

__device__ inline void edge_endpoints(const int* __restrict__ ei, int e, int& s, int& d) {
    if (e < NEDGES) { s = ei[e]; d = ei[NEDGES + e]; }
    else            { s = d = e - NEDGES; }   // appended self-loops
}

__global__ void edge_max_kernel(const int* __restrict__ ei, const float* __restrict__ asrc,
                                const float* __restrict__ adst, float* __restrict__ mmax, int H) {
    int e = blockIdx.x * blockDim.x + threadIdx.x;
    if (e >= ETOT) return;
    int s, d; edge_endpoints(ei, e, s, d);
    for (int hh = 0; hh < H; hh++) {
        float v = asrc[s * H + hh] + adst[d * H + hh];
        v = v > 0.f ? v : NEG_SLOPE * v;
        atomicMaxFloat(&mmax[d * H + hh], v);
    }
}

__global__ void edge_sum_kernel(const int* __restrict__ ei, const float* __restrict__ asrc,
                                const float* __restrict__ adst, const float* __restrict__ mmax,
                                float* __restrict__ denom, int H) {
    int e = blockIdx.x * blockDim.x + threadIdx.x;
    if (e >= ETOT) return;
    int s, d; edge_endpoints(ei, e, s, d);
    for (int hh = 0; hh < H; hh++) {
        float v = asrc[s * H + hh] + adst[d * H + hh];
        v = v > 0.f ? v : NEG_SLOPE * v;
        atomicAdd(&denom[d * H + hh], __expf(v - mmax[d * H + hh]));
    }
}

// one wave32 per edge: coalesced gather of h[src] rows, f32 atomic scatter into out[dst]
__global__ void aggregate_kernel(const int* __restrict__ ei, const float* __restrict__ hfeat,
                                 const float* __restrict__ asrc, const float* __restrict__ adst,
                                 const float* __restrict__ mmax, const float* __restrict__ denom,
                                 float* __restrict__ out, int H, int F) {
    long gid = (long)blockIdx.x * blockDim.x + threadIdx.x;
    int e = (int)(gid >> 5);
    int lane = threadIdx.x & 31;
    if (e >= ETOT) return;
    int s, d; edge_endpoints(ei, e, s, d);

    float a = 0.f;
    if (lane < H) {
        float v = asrc[s * H + lane] + adst[d * H + lane];
        v = v > 0.f ? v : NEG_SLOPE * v;
        a = __expf(v - mmax[d * H + lane]) / denom[d * H + lane];
    }
    const float* hs = hfeat + (long)s * F;
    float* od = out + (long)d * F;
    for (int f = lane; f < F; f += 32) {
        int head = (H == 1) ? 0 : (f >> 6);        // C == 64 when H == 8
        float alpha = __shfl(a, head);
        atomicAdd(&od[f], hs[f] * alpha);
    }
}

// ---------------- epilogues ----------------

__global__ void bias_elu_bf16_kernel(const float* __restrict__ in, const float* __restrict__ b,
                                     bf16* __restrict__ out, int F, long total) {
    long i = (long)blockIdx.x * blockDim.x + threadIdx.x;
    if (i >= total) return;
    int f = (int)(i % F);
    float v = in[i] + b[f];
    v = v > 0.f ? v : (__expf(v) - 1.f);   // ELU
    out[i] = (bf16)v;
}

__global__ void lsm_kernel(const float* __restrict__ in, const float* __restrict__ b,
                           float* __restrict__ out) {
    int n = blockIdx.x * blockDim.x + threadIdx.x;
    if (n >= NNODES) return;
    float buf[NOUT];
    float mx = -3.0e38f;
    for (int c = 0; c < NOUT; c++) { buf[c] = in[n * NOUT + c] + b[c]; mx = fmaxf(mx, buf[c]); }
    float ssum = 0.f;
    for (int c = 0; c < NOUT; c++) ssum += __expf(buf[c] - mx);
    float ls = __logf(ssum);
    for (int c = 0; c < NOUT; c++) out[n * NOUT + c] = buf[c] - mx - ls;
}

// ---------------- orchestration ----------------

extern "C" void kernel_launch(void* const* d_in, const int* in_sizes, int n_in,
                              void* d_out, int out_size, void* d_ws, size_t ws_size,
                              hipStream_t stream) {
    (void)in_sizes; (void)n_in; (void)out_size; (void)ws_size;
    const float* x   = (const float*)d_in[0];
    const int*   ei  = (const int*)d_in[1];
    const float* W[3]  = {(const float*)d_in[2], (const float*)d_in[6],  (const float*)d_in[10]};
    const float* As[3] = {(const float*)d_in[3], (const float*)d_in[7],  (const float*)d_in[11]};
    const float* Ad[3] = {(const float*)d_in[4], (const float*)d_in[8],  (const float*)d_in[12]};
    const float* Bi[3] = {(const float*)d_in[5], (const float*)d_in[9],  (const float*)d_in[13]};
    float* outp = (float*)d_out;

    char* base = (char*)d_ws;
    size_t off = 0;
    auto carve = [&](size_t bytes) -> void* {
        void* p = base + off;
        off = (off + bytes + 255) & ~(size_t)255;
        return p;
    };
    bf16*  xb  = (bf16*) carve((size_t)NNODES * FHID * 2);   // bf16 layer input (FIN==FHID)
    bf16*  wt  = (bf16*) carve((size_t)512 * 512 * 2);       // transposed bf16 weights
    float* hb  = (float*)carve((size_t)NNODES * FHID * 4);   // GEMM output h
    float* ob  = (float*)carve((size_t)NNODES * FHID * 4);   // aggregated output
    float* asb = (float*)carve((size_t)NNODES * HEADS * 4);
    float* adb = (float*)carve((size_t)NNODES * HEADS * 4);
    float* mm  = (float*)carve((size_t)NNODES * HEADS * 4);
    float* dn  = (float*)carve((size_t)NNODES * HEADS * 4);

    const int TB = 256;
    auto nb = [](long n, int tb) { return (int)((n + tb - 1) / tb); };

    cvt_bf16_kernel<<<nb((long)NNODES * FIN, TB), TB, 0, stream>>>(x, xb, (long)NNODES * FIN);

    const int Kdim[3]  = {FIN,  FHID, FHID};
    const int Nfdim[3] = {FHID, FHID, NOUT};
    const int Hh[3]    = {HEADS, HEADS, 1};

    for (int l = 0; l < 3; l++) {
        const int K = Kdim[l], Nf = Nfdim[l];
        const int NfPad = ((Nf + 15) / 16) * 16;
        const int NT = NfPad / 16;
        const int H = Hh[l];
        const int F = Nf == NOUT ? NOUT : FHID;   // H*C

        transpose_bf16_kernel<<<nb((long)NfPad * K, TB), TB, 0, stream>>>(W[l], wt, K, Nf, NfPad);
        if (NT >= 32)
            gemm_wmma_kernel<4><<<NNODES / 16, 256, 0, stream>>>(xb, wt, hb, K, Nf, NT);
        else
            gemm_wmma_kernel<1><<<NNODES / 16, 256, 0, stream>>>(xb, wt, hb, K, Nf, NT);
        alpha_kernel<<<nb((long)NNODES * H, TB), TB, 0, stream>>>(hb, As[l], Ad[l], asb, adb, H, F / H);

        fill_f32_kernel<<<nb((long)NNODES * H, TB), TB, 0, stream>>>(mm, -3.0e38f, (long)NNODES * H);
        fill_f32_kernel<<<nb((long)NNODES * H, TB), TB, 0, stream>>>(dn, 0.f, (long)NNODES * H);
        fill_f32_kernel<<<nb((long)NNODES * F, TB), TB, 0, stream>>>(ob, 0.f, (long)NNODES * F);

        edge_max_kernel<<<nb(ETOT, TB), TB, 0, stream>>>(ei, asb, adb, mm, H);
        edge_sum_kernel<<<nb(ETOT, TB), TB, 0, stream>>>(ei, asb, adb, mm, dn, H);
        aggregate_kernel<<<nb((long)ETOT * 32, TB), TB, 0, stream>>>(ei, hb, asb, adb, mm, dn, ob, H, F);

        if (l < 2) {
            bias_elu_bf16_kernel<<<nb((long)NNODES * F, TB), TB, 0, stream>>>(
                ob, Bi[l], xb, F, (long)NNODES * F);
        } else {
            lsm_kernel<<<nb(NNODES, TB), TB, 0, stream>>>(ob, Bi[l], outp);
        }
    }
}
